// BitLinear15b_57836029608232
// MI455X (gfx1250) — 8-bit matrix operand order repaired and checked by reading, compiled, not run on hardware
//
#include <hip/hip_runtime.h>

typedef int  v8i __attribute__((ext_vector_type(8)));

#define KDIM 2048
#define NDIM 2048
#define EPSG 1e-6f

__global__ void zero_scalar_kernel(float* p) {
    if (threadIdx.x == 0 && blockIdx.x == 0) *p = 0.0f;
}

__global__ __launch_bounds__(256) void abssum_kernel(const float* __restrict__ W,
                                                     int n, float* __restrict__ out) {
    __shared__ float sdata[256];
    float s = 0.0f;
    for (int i = blockIdx.x * blockDim.x + threadIdx.x; i < n;
         i += gridDim.x * blockDim.x)
        s += fabsf(W[i]);
    sdata[threadIdx.x] = s;
    __syncthreads();
    for (int off = 128; off > 0; off >>= 1) {
        if (threadIdx.x < off) sdata[threadIdx.x] += sdata[threadIdx.x + off];
        __syncthreads();
    }
    if (threadIdx.x == 0) atomicAdd(out, sdata[0]);
}

__global__ __launch_bounds__(256) void quantw_kernel(const float* __restrict__ W,
                                                     const float* __restrict__ sumAbs,
                                                     signed char* __restrict__ Wq,
                                                     int n) {
    const float gamma = (*sumAbs) / (float)n + EPSG;
    const float thr   = 0.5f * gamma;
    const int base = (blockIdx.x * blockDim.x + threadIdx.x) * 4;
    if (base >= n) return;
    const float4 w = *(const float4*)(W + base);
    char4 q;
    q.x = (fabsf(w.x) > thr) ? ((w.x > 0.0f) ? 1 : -1) : 0;
    q.y = (fabsf(w.y) > thr) ? ((w.y > 0.0f) ? 1 : -1) : 0;
    q.z = (fabsf(w.z) > thr) ? ((w.z > 0.0f) ? 1 : -1) : 0;
    q.w = (fabsf(w.w) > thr) ? ((w.w > 0.0f) ? 1 : -1) : 0;
    *(char4*)(Wq + base) = q;
}

__global__ __launch_bounds__(256) void signx_kernel(const float* __restrict__ X,
                                                    signed char* __restrict__ Xs,
                                                    int n) {
    const int base = (blockIdx.x * blockDim.x + threadIdx.x) * 4;
    if (base >= n) return;
    const float4 x = *(const float4*)(X + base);
    char4 s;
    s.x = (x.x > 0.0f) ? 1 : ((x.x < 0.0f) ? -1 : 0);
    s.y = (x.y > 0.0f) ? 1 : ((x.y < 0.0f) ? -1 : 0);
    s.z = (x.z > 0.0f) ? 1 : ((x.z < 0.0f) ? -1 : 0);
    s.w = (x.w > 0.0f) ? 1 : ((x.w < 0.0f) ? -1 : 0);
    *(char4*)(Xs + base) = s;
}

__device__ __forceinline__ v8i load_a_frag(const signed char* p) {
    const int2* q = (const int2*)p;
    int2 t0 = q[0], t1 = q[2], t2 = q[4], t3 = q[6];
    v8i a;
    a[0] = t0.x; a[1] = t0.y; a[2] = t1.x; a[3] = t1.y;
    a[4] = t2.x; a[5] = t2.y; a[6] = t3.x; a[7] = t3.y;
    return a;
}

__device__ __forceinline__ v8i load_b_frag(const signed char* p) {
    const int2* q = (const int2*)p;
    int2 t0 = q[0], t1 = q[2], t2 = q[4], t3 = q[6];
    v8i b;
    b[0] = t0.x; b[1] = t0.y; b[2] = t1.x; b[3] = t1.y;
    b[4] = t2.x; b[5] = t2.y; b[6] = t3.x; b[7] = t3.y;
    return b;
}

__global__ __launch_bounds__(256) void gemm_iu8_kernel(
        const signed char* __restrict__ A,
        const signed char* __restrict__ B,
        float* __restrict__ C,
        int M) {
    const int lane = threadIdx.x & 31;
    const int wave = threadIdx.x >> 5;
    const int lo   = lane & 15;
    const int hi   = lane >> 4;
    const int wm   = wave & 3;
    const int wn   = wave >> 2;
    const int mBase = blockIdx.y * 128 + wm * 32;
    const int nBase = blockIdx.x * 128 + wn * 64;

    v8i acc[2][4];
#pragma unroll
    for (int i = 0; i < 2; ++i)
#pragma unroll
        for (int j = 0; j < 4; ++j)
            acc[i][j] = (v8i)0;

    const signed char* aP0 = A + (size_t)(mBase + lo) * KDIM + hi * 8;
    const signed char* aP1 = aP0 + (size_t)16 * KDIM;
    const signed char* bP[4];
#pragma unroll
    for (int j = 0; j < 4; ++j)
        bP[j] = B + (size_t)(nBase + j * 16 + lo) * KDIM + hi * 8;

    for (int k0 = 0; k0 < KDIM; k0 += 64) {
        v8i a0 = load_a_frag(aP0 + k0);
        v8i a1 = load_a_frag(aP1 + k0);
        v8i b0 = load_b_frag(bP[0] + k0);
        v8i b1 = load_b_frag(bP[1] + k0);
        v8i b2 = load_b_frag(bP[2] + k0);
        v8i b3 = load_b_frag(bP[3] + k0);

        acc[0][0] = __builtin_amdgcn_wmma_i32_16x16x64_iu8(true, a0, true, b0, acc[0][0], false, false);
        acc[0][1] = __builtin_amdgcn_wmma_i32_16x16x64_iu8(true, a0, true, b1, acc[0][1], false, false);
        acc[0][2] = __builtin_amdgcn_wmma_i32_16x16x64_iu8(true, a0, true, b2, acc[0][2], false, false);
        acc[0][3] = __builtin_amdgcn_wmma_i32_16x16x64_iu8(true, a0, true, b3, acc[0][3], false, false);
        acc[1][0] = __builtin_amdgcn_wmma_i32_16x16x64_iu8(true, a1, true, b0, acc[1][0], false, false);
        acc[1][1] = __builtin_amdgcn_wmma_i32_16x16x64_iu8(true, a1, true, b1, acc[1][1], false, false);
        acc[1][2] = __builtin_amdgcn_wmma_i32_16x16x64_iu8(true, a1, true, b2, acc[1][2], false, false);
        acc[1][3] = __builtin_amdgcn_wmma_i32_16x16x64_iu8(true, a1, true, b3, acc[1][3], false, false);
    }

#pragma unroll
    for (int i = 0; i < 2; ++i) {
#pragma unroll
        for (int j = 0; j < 4; ++j) {
            float* cp = C + (size_t)(mBase + i * 16 + hi * 8) * NDIM
                          + (nBase + j * 16 + lo);
#pragma unroll
            for (int r = 0; r < 8; ++r)
                cp[(size_t)r * NDIM] = (float)acc[i][j][r];
        }
    }
}

extern "C" void kernel_launch(void* const* d_in, const int* in_sizes, int n_in,
                              void* d_out, int out_size, void* d_ws, size_t ws_size,
                              hipStream_t stream) {
    const float* x = (const float*)d_in[0];
    const float* w = (const float*)d_in[1];
    float* out = (float*)d_out;

    const int nW = in_sizes[1];
    const int nX = in_sizes[0];
    const int M  = nX / KDIM;

    char* ws = (char*)d_ws;
    float*       gammaSum = (float*)ws;
    signed char* Wq       = (signed char*)(ws + 256);
    signed char* Xs       = (signed char*)(ws + 256 + (size_t)NDIM * KDIM);

    zero_scalar_kernel<<<1, 64, 0, stream>>>(gammaSum);

    abssum_kernel<<<1024, 256, 0, stream>>>(w, nW, gammaSum);

    quantw_kernel<<<(nW / 4 + 255) / 256, 256, 0, stream>>>(w, gammaSum, Wq, nW);

    signx_kernel<<<(nX / 4 + 255) / 256, 256, 0, stream>>>(x, Xs, nX);

    dim3 grid(NDIM / 128, M / 128);
    gemm_iu8_kernel<<<grid, 256, 0, stream>>>(Xs, Wq, out, M);
}
